// CSPN_59442347377426
// MI455X (gfx1250) — matrix operand order repaired and checked by reading, not compiled, not run on hardware
//
#include <hip/hip_runtime.h>
#include <math.h>

typedef __attribute__((ext_vector_type(16))) _Float16 v16h;
typedef __attribute__((ext_vector_type(8)))  float    v8f;

// Pointer types for the async-to-LDS builtin:
// param0: 'int __attribute__((vector_size(16))) __device__ *'  (addrspace(1))
// param1: same pointee in addrspace(3)
typedef int v4i_gcc __attribute__((vector_size(16)));
typedef __attribute__((address_space(1))) v4i_gcc as1_v4i;
typedef __attribute__((address_space(3))) v4i_gcc as3_v4i;

#if defined(__HIP_DEVICE_COMPILE__) && __has_builtin(__builtin_amdgcn_global_load_async_to_lds_b128)
#define ASYNC_LDS 1
#else
#define ASYNC_LDS 0
#endif

// Problem constants (match reference): B=4096, K=32, NUM_SUMS=32, D=64.
// gate_params layout: (B, 1024, 32) row-major -> row c is 32 contiguous floats.

__global__ __launch_bounds__(256) void cspn_fused_kernel(
    const float* __restrict__ x,       // (B, 64)
    const float* __restrict__ means1,  // (B, 1024)
    const float* __restrict__ means2,  // (B, 1024)
    const float* __restrict__ gate,    // (B, 1024, 32)
    float* __restrict__ out)           // (B, 32)
{
  __shared__ float sh_x[64];
  __shared__ float sh_lp[2][32];
  __shared__ float sh_E1[32];
  __shared__ float sh_E2[32];
  __shared__ float sh_P[1024];
  __shared__ float sh_red1[32][32];
  __shared__ float sh_red2[32][32];
  __shared__ float sh_gmax[32];
  __shared__ float sh_scal[3];          // m1, m2, mp = m1+m2
  extern __shared__ float4 sh_stage[];  // 8192 float4 = 128 KB gate slice

  const int b    = blockIdx.x;
  const int tid  = threadIdx.x;
  const int lane = tid & 31;
  const int wid  = tid >> 5;

  // ---- Phase 0: kick off async staging of the whole gate slice --------
  // 256 threads x 32 iters x 16B = 128 KB. Fire-and-forget (ASYNCcnt),
  // overlapped with the WMMA leaf computation below.
  const float4* g4 = (const float4*)(gate + (size_t)b * 32768);
#if ASYNC_LDS
#pragma unroll
  for (int k = 0; k < 32; ++k) {
    const int idx = (k << 8) + tid;
    __builtin_amdgcn_global_load_async_to_lds_b128(
        (as1_v4i*)(v4i_gcc*)(g4 + idx),
        (as3_v4i*)(v4i_gcc*)(sh_stage + idx),
        0, 0);
  }
#else
  for (int k = 0; k < 32; ++k) {
    const int idx = (k << 8) + tid;
    sh_stage[idx] = g4[idx];
  }
#endif

  if (tid < 64) sh_x[tid] = x[(size_t)b * 64 + tid];
  __syncthreads();

  // ---- Phase 1: leaf log-probs via WMMA -------------------------------
  // lp[i] = -0.5 * sum_j (x[j]-mu[i,j])^2 - 16*ln(2pi)
  // = -0.5 * diag(Diff * Diff^T) - const, computed with
  // v_wmma_f32_16x16x32_f16 on 16-row tiles. Waves 0..3: (leaf, tile).
  if (wid < 4) {
    const int leaf = wid & 1;
    const int nb   = (wid >> 1) << 4;   // tile base row: 0 or 16
    const float* mu = (leaf ? means2 : means1) + (size_t)b * 1024;
    const float* xl = sh_x + leaf * 32;

    // A-matrix (16x32 f16): lane m = lane&15 holds row (nb+m); half h=lane>>4
    // VGPR v in 0..3 -> K = 2v+8h{,+1}; v in 4..7 -> K = 16+2(v-4)+8h{,+1}
    const int mrow = nb + (lane & 15);
    const int h    = lane >> 4;
    v16h a;
#pragma unroll
    for (int v = 0; v < 8; ++v) {
      const int kb = (v < 4) ? (2 * v + 8 * h) : (16 + 2 * (v - 4) + 8 * h);
      a[2 * v]     = (_Float16)(xl[kb]     - mu[mrow * 32 + kb]);
      a[2 * v + 1] = (_Float16)(xl[kb + 1] - mu[mrow * 32 + kb + 1]);
    }
    // B-matrix (32x16 f16): lane n = lane&15 holds column n; K order same as A.
    // B[k][n] = Diff[nb+n][k]
    v16h bm;
#pragma unroll
    for (int n = 0; n < 16; ++n) {
      const int kk = (n < 8) ? (n + 8 * h) : (16 + (n - 8) + 8 * h);
      bm[n] = (_Float16)(xl[kk] - mu[mrow * 32 + kk]);
    }
    v8f c = {};
    c = __builtin_amdgcn_wmma_f32_16x16x32_f16(false, a, false, bm,
                                               (short)0, c, false, false);
    // C/D layout: lanes 0-15: N=lane, M=r; lanes 16-31: N=lane-16, M=r+8.
    // Diagonal M==N lives in lanes 0..7 (r=lane) and 24..31 (r=lane-24).
    // Select-accumulate (v_cndmask) then one predicated store.
    const float cst = 29.40603306f; // 16*ln(2*pi)
    float diag = 0.f;
#pragma unroll
    for (int r = 0; r < 8; ++r) {
      const int m = (lane < 16) ? r : (r + 8);
      diag += (m == (lane & 15)) ? c[r] : 0.f;
    }
    if (lane < 8 || lane >= 24)
      sh_lp[leaf][nb + (lane & 15)] = -0.5f * diag - cst;
  }
  __syncthreads();

  // ---- Phase 2: prod-exponentials P[c] = exp(prod[c] - max prod) ------
  if (tid == 0) {
    float m1 = sh_lp[0][0], m2 = sh_lp[1][0];
    for (int i = 1; i < 32; ++i) {
      m1 = fmaxf(m1, sh_lp[0][i]);
      m2 = fmaxf(m2, sh_lp[1][i]);
    }
    sh_scal[0] = m1; sh_scal[1] = m2; sh_scal[2] = m1 + m2;
  }
  __syncthreads();
  if (tid < 32)      sh_E1[tid]      = __expf(sh_lp[0][tid]      - sh_scal[0]);
  else if (tid < 64) sh_E2[tid - 32] = __expf(sh_lp[1][tid - 32] - sh_scal[1]);
  __syncthreads();
  for (int cc = tid; cc < 1024; cc += 256)
    sh_P[cc] = sh_E1[cc & 31] * sh_E2[cc >> 5];  // prod[c]=lp1[c&31]+lp2[c>>5]

  // ---- Wait for the async gate staging, then barrier ------------------
#if ASYNC_LDS
#if __has_builtin(__builtin_amdgcn_s_wait_asynccnt)
  __builtin_amdgcn_s_wait_asynccnt(0);
#else
  asm volatile("s_wait_asynccnt 0" ::: "memory");
#endif
#endif
  __syncthreads();

  // ---- Phase 3: both logsumexp passes entirely out of LDS -------------
  // Lane owns 4 adjacent s columns: si = lane&7 -> s = 4*si..4*si+3.
  // A wave covers 4 rows per step (coff = lane>>3); 8 waves x 32 steps = 1024 rows.
  const int si   = lane & 7;
  const int coff = lane >> 3;

  // Pass 1: column maxes (pure v_max, no exp)
  float4 mx = make_float4(-INFINITY, -INFINITY, -INFINITY, -INFINITY);
#pragma unroll 8
  for (int t = 0; t < 32; ++t) {
    const int c = ((t << 3) + wid) * 4 + coff;
    const float4 v = sh_stage[c * 8 + si];
    mx.x = fmaxf(mx.x, v.x); mx.y = fmaxf(mx.y, v.y);
    mx.z = fmaxf(mx.z, v.z); mx.w = fmaxf(mx.w, v.w);
  }
  {
    const int r = wid * 4 + coff;
    sh_red1[r][4 * si + 0] = mx.x; sh_red1[r][4 * si + 1] = mx.y;
    sh_red1[r][4 * si + 2] = mx.z; sh_red1[r][4 * si + 3] = mx.w;
  }
  __syncthreads();
  if (tid < 32) {
    float m = sh_red1[0][tid];
    for (int r = 1; r < 32; ++r) m = fmaxf(m, sh_red1[r][tid]);
    sh_gmax[tid] = m;
  }
  __syncthreads();

  // Pass 2: 1 exp + 2 fma per gate element, from LDS.
  float4 gm;
  gm.x = sh_gmax[4 * si + 0]; gm.y = sh_gmax[4 * si + 1];
  gm.z = sh_gmax[4 * si + 2]; gm.w = sh_gmax[4 * si + 3];
  float4 t1 = make_float4(0.f, 0.f, 0.f, 0.f);
  float4 t2 = make_float4(0.f, 0.f, 0.f, 0.f);
#pragma unroll 8
  for (int t = 0; t < 32; ++t) {
    const int c = ((t << 3) + wid) * 4 + coff;
    const float4 v = sh_stage[c * 8 + si];
    const float pc = sh_P[c];
    float e;
    e = __expf(v.x - gm.x); t2.x += e; t1.x = fmaf(e, pc, t1.x);
    e = __expf(v.y - gm.y); t2.y += e; t1.y = fmaf(e, pc, t1.y);
    e = __expf(v.z - gm.z); t2.z += e; t1.z = fmaf(e, pc, t1.z);
    e = __expf(v.w - gm.w); t2.w += e; t1.w = fmaf(e, pc, t1.w);
  }
  {
    const int r = wid * 4 + coff;
    sh_red1[r][4 * si + 0] = t1.x; sh_red2[r][4 * si + 0] = t2.x;
    sh_red1[r][4 * si + 1] = t1.y; sh_red2[r][4 * si + 1] = t2.y;
    sh_red1[r][4 * si + 2] = t1.z; sh_red2[r][4 * si + 2] = t2.z;
    sh_red1[r][4 * si + 3] = t1.w; sh_red2[r][4 * si + 3] = t2.w;
  }
  __syncthreads();

  // ---- Phase 4: final reduce + log ------------------------------------
  // out[b,s] = log(sum e*P) - log(sum e) + (m1+m2)
  if (tid < 32) {
    float a1 = 0.f, a2 = 0.f;
    for (int r = 0; r < 32; ++r) { a1 += sh_red1[r][tid]; a2 += sh_red2[r][tid]; }
    out[(size_t)b * 32 + tid] = __logf(a1) - __logf(a2) + sh_scal[2];
  }
}

extern "C" void kernel_launch(void* const* d_in, const int* in_sizes, int n_in,
                              void* d_out, int out_size, void* d_ws, size_t ws_size,
                              hipStream_t stream) {
  const float* x      = (const float*)d_in[0];
  const float* means1 = (const float*)d_in[1];
  const float* means2 = (const float*)d_in[2];
  const float* gate   = (const float*)d_in[3];
  // d_in[4]/d_in[5] are scope1/scope2 (fixed arange(0,32)/arange(32,64)); baked in.
  float* out = (float*)d_out;

  const int B = in_sizes[0] / 64;  // 4096
  const size_t dyn_lds = 32768 * sizeof(float);  // 128 KB gate slice
  dim3 grid(B), block(256);
  hipLaunchKernelGGL(cspn_fused_kernel, grid, block, dyn_lds, stream,
                     x, means1, means2, gate, out);
}
